// TrajVAE_84361747628758
// MI455X (gfx1250) — compile-verified, matrix operand order checked by reading, not run on hardware
//
#include <hip/hip_runtime.h>
#include <hip/hip_bf16.h>

// ---------------- problem constants ----------------
constexpr int cB   = 8192;
constexpr int cT   = 30;
constexpr int cH   = 256;   // hidden
constexpr int cG4  = 1024;  // 4*H gate width
constexpr float cDT = 0.03f;
constexpr int cROWS = 32;   // batch rows per block (2 row-tiles of 16)

typedef __attribute__((ext_vector_type(16))) _Float16 v16h;
typedef __attribute__((ext_vector_type(8)))  _Float16 h8;
typedef __attribute__((ext_vector_type(8)))  float    v8f;

// Load a 16-element f16 fragment (A or B operand) for wmma_f32_16x16x32_f16.
// Per-lane layout: chunk0 = K[k0*32+koff .. +8), chunk1 = K[k0*32+koff+16 .. +8)
// where koff = (lane>>4)*8 and the row/col index is (lane&15).
__device__ __forceinline__ v16h load_frag(const _Float16* __restrict__ p, int koff) {
    h8 lo = *reinterpret_cast<const h8*>(p + koff);
    h8 hi = *reinterpret_cast<const h8*>(p + koff + 16);
    v16h a;
#pragma unroll
    for (int i = 0; i < 8; ++i) { a[i] = lo[i]; a[8 + i] = hi[i]; }
    return a;
}

__device__ __forceinline__ float sigmoidf_(float x) { return 1.0f / (1.0f + __expf(-x)); }

// ---------------- prep kernels ----------------
__global__ void fuse_in_kernel(const float* __restrict__ embW, const float* __restrict__ embB,
                               const float* __restrict__ Wih, const float* __restrict__ bih,
                               const float* __restrict__ bhh,
                               float* __restrict__ Wf, float* __restrict__ biasTot) {
    int i = blockIdx.x * blockDim.x + threadIdx.x;
    if (i < 4096) {
        int k = i >> 10, n = i & 1023;
        float s = 0.f;
        for (int e = 0; e < 128; ++e) s += embW[k * 128 + e] * Wih[e * 1024 + n];
        Wf[i] = s;
    } else if (i < 5120) {
        int n = i - 4096;
        float s = bih[n] + bhh[n];
        for (int e = 0; e < 128; ++e) s += embB[e] * Wih[e * 1024 + n];
        biasTot[n] = s;
    }
}

// dst[n][k] = (f16)src[k][n]   (src is K x N row-major)
__global__ void transpose_f16_kernel(const float* __restrict__ src, _Float16* __restrict__ dst,
                                     int K, int N) {
    int i = blockIdx.x * blockDim.x + threadIdx.x;
    if (i < K * N) {
        int k = i / N, n = i - k * N;
        dst[n * K + k] = (_Float16)src[i];
    }
}

__global__ void zero_hc_kernel(_Float16* __restrict__ h, float* __restrict__ c, int n) {
    int i = blockIdx.x * blockDim.x + threadIdx.x;
    if (i < n) { h[i] = (_Float16)0.0f; c[i] = 0.0f; }
}

__global__ void copy_state_kernel(const float* __restrict__ src, float* __restrict__ dst, int n) {
    int i = blockIdx.x * blockDim.x + threadIdx.x;
    if (i < n) dst[i] = src[i];
}

__global__ void reparam_kernel(const float* __restrict__ mu, const float* __restrict__ lv,
                               const float* __restrict__ eps, _Float16* __restrict__ z, int n) {
    int i = blockIdx.x * blockDim.x + threadIdx.x;
    if (i < n) z[i] = (_Float16)tanhf(eps[i] * __expf(0.5f * lv[i]) + mu[i]);
}

// ---------------- fused LSTM step (WMMA) ----------------
// Block: 32 batch rows x 1024 gates. 8 waves; wave w owns gate cols {q*256+32w..+32}
// for all four quadrants q -> LSTM pointwise is lane-local (no LDS gate exchange).
// Each loaded B (weight) fragment is amortized over 2 row-tiles to halve L2 weight
// traffic (the dominant cost: weights >> activations here).
template <int IS_DEC>
__global__ __launch_bounds__(256) void lstm_step_kernel(
    _Float16* __restrict__ hbuf,            // [B][256] f16, in/out (in place)
    float* __restrict__ cbuf,               // [B][256] f32, in/out
    const _Float16* __restrict__ Wt,        // [1024][256] f16 (Whh transposed)
    const float* __restrict__ Wf,           // [4][1024] fused input weights
    const float* __restrict__ biasTot,      // [1024]
    const float* __restrict__ traj,         // expert_traj (enc) else unused
    float* __restrict__ state,              // [B][4] (dec) else unused
    const float* __restrict__ ctrlW,        // [256][2] (dec)
    const float* __restrict__ ctrlB,        // [2] (dec)
    float* __restrict__ recons,             // [B][T][4] out (dec)
    int t) {
    __shared__ float xs[cROWS][4];
    __shared__ float htile[IS_DEC ? cROWS : 1][IS_DEC ? (cH + 1) : 1];  // +1: bank-conflict pad

    const int tid  = threadIdx.x;
    const int wv   = tid >> 5;
    const int lane = tid & 31;
    const int m0   = blockIdx.x * cROWS;

    // stage x tile (rel input for enc, current state for dec)
    if (tid < cROWS * 4) {
        int mi = tid >> 2, k = tid & 3;
        int b = m0 + mi;
        float xv;
        if (IS_DEC) {
            xv = state[b * 4 + k];
        } else {
            float cur = traj[(b * cT + t) * 4 + k];
            xv = (k < 2 && t > 0) ? (cur - traj[(b * cT + t - 1) * 4 + k]) : cur;
        }
        xs[mi][k] = xv;
    }
    __syncthreads();

    // ---- A fragments: h_prev rows m0..m0+31 (2 row-tiles), K=256 ----
    const int koff = (lane >> 4) * 8;
    v16h af[2][8];
#pragma unroll
    for (int r = 0; r < 2; ++r) {
        const int rowA = m0 + r * 16 + (lane & 15);
#pragma unroll
        for (int k0 = 0; k0 < 8; ++k0)
            af[r][k0] = load_frag(hbuf + rowA * cH + k0 * 32, koff);
    }

    // ---- 8 N-tiles x 2 row-tiles per wave ----
    v8f acc[2][8];
#pragma unroll
    for (int r = 0; r < 2; ++r)
#pragma unroll
        for (int i = 0; i < 8; ++i)
#pragma unroll
            for (int v = 0; v < 8; ++v) acc[r][i][v] = 0.0f;

#pragma unroll
    for (int q = 0; q < 4; ++q) {
#pragma unroll
        for (int cc = 0; cc < 2; ++cc) {
            const int n0 = q * cH + wv * 32 + cc * 16;
            const _Float16* wrow = Wt + (n0 + (lane & 15)) * cH;
#pragma unroll
            for (int k0 = 0; k0 < 8; ++k0) {
                v16h bf = load_frag(wrow + k0 * 32, koff);   // loaded once,
#pragma unroll
                for (int r = 0; r < 2; ++r)                  // used twice
                    acc[r][q * 2 + cc] = __builtin_amdgcn_wmma_f32_16x16x32_f16(
                        false, af[r][k0], false, bf, (short)0, acc[r][q * 2 + cc], false, false);
            }
        }
    }

    // all waves must be done reading h_prev before we overwrite it
    __syncthreads();

    // ---- lane-local LSTM pointwise ----
    const int nloc = lane & 15;
    const int mb   = (lane >> 4) * 8;
#pragma unroll
    for (int cc = 0; cc < 2; ++cc) {
        const int nc = wv * 32 + cc * 16 + nloc;  // cell column in [0,256)
        float bq[4], w0[4], w1[4], w2[4], w3[4];
#pragma unroll
        for (int q = 0; q < 4; ++q) {
            int n = q * cH + nc;
            bq[q] = biasTot[n];
            w0[q] = Wf[n]; w1[q] = Wf[1024 + n]; w2[q] = Wf[2048 + n]; w3[q] = Wf[3072 + n];
        }
#pragma unroll
        for (int r = 0; r < 2; ++r) {
#pragma unroll
            for (int v = 0; v < 8; ++v) {
                const int mi = r * 16 + mb + v;
                const int gm = m0 + mi;
                const float x0 = xs[mi][0], x1 = xs[mi][1], x2 = xs[mi][2], x3 = xs[mi][3];
                float g[4];
#pragma unroll
                for (int q = 0; q < 4; ++q)
                    g[q] = acc[r][q * 2 + cc][v] + bq[q]
                         + x0 * w0[q] + x1 * w1[q] + x2 * w2[q] + x3 * w3[q];
                const float ig = sigmoidf_(g[0]);
                const float fg = sigmoidf_(g[1]);
                const float gg = tanhf(g[2]);
                const float og = sigmoidf_(g[3]);
                const float cold = cbuf[gm * cH + nc];
                const float cnew = fg * cold + ig * gg;
                cbuf[gm * cH + nc] = cnew;
                const float hnew = og * tanhf(cnew);
                hbuf[gm * cH + nc] = (_Float16)hnew;
                if (IS_DEC) htile[mi][nc] = hnew;
            }
        }
    }

    if (IS_DEC) {
        __syncthreads();
        if (tid < cROWS) {
            const int mi = tid, gm = m0 + mi;
            float pedal = ctrlB[0], steer = ctrlB[1];
            for (int k = 0; k < cH; ++k) {
                const float hv = htile[mi][k];
                pedal += hv * ctrlW[k * 2 + 0];
                steer += hv * ctrlW[k * 2 + 1];
            }
            steer = fminf(0.5f, fmaxf(-0.5f, steer));
            const float sx = xs[mi][0], sy = xs[mi][1], psi = xs[mi][2], vel = xs[mi][3];
            const float v1 = fminf(10.0f, fmaxf(0.0f, vel + pedal * cDT));
            float pd = vel * tanf(steer) / 2.5f;
            pd = fminf(3.14f / 2.0f, fmaxf(-3.14f / 2.0f, pd));
            const float o0 = sx + v1 * cosf(psi) * cDT;
            const float o1 = sy + v1 * sinf(psi) * cDT;
            const float o2 = psi + pd * cDT;
            float out4[4] = {o0, o1, o2, v1};
            for (int j = 0; j < 4; ++j) {
                state[gm * 4 + j] = out4[j];
                recons[(gm * cT + t) * 4 + j] = out4[j];
            }
        }
    }
}

// ---------------- WMMA GEMM + activation (MLP layers / dec_init) ----------------
// out = act(A[B x 256] @ W[256 x 256] + bias); W pre-transposed to [N][K] f16.
// 32 rows per block; each weight fragment feeds 2 row-tiles.
__global__ __launch_bounds__(256) void gemm_act_kernel(
    const _Float16* __restrict__ A, const _Float16* __restrict__ Wt,
    const float* __restrict__ bias,
    _Float16* __restrict__ out16, float* __restrict__ out32, int act) {
    const int tid = threadIdx.x, wv = tid >> 5, lane = tid & 31;
    const int m0 = blockIdx.x * cROWS;
    const int koff = (lane >> 4) * 8;

    v16h af[2][8];
#pragma unroll
    for (int r = 0; r < 2; ++r) {
        const int rowA = m0 + r * 16 + (lane & 15);
#pragma unroll
        for (int k0 = 0; k0 < 8; ++k0)
            af[r][k0] = load_frag(A + rowA * cH + k0 * 32, koff);
    }

    v8f acc[2][2];
#pragma unroll
    for (int r = 0; r < 2; ++r)
#pragma unroll
        for (int i = 0; i < 2; ++i)
#pragma unroll
            for (int v = 0; v < 8; ++v) acc[r][i][v] = 0.0f;

#pragma unroll
    for (int cc = 0; cc < 2; ++cc) {
        const int n0 = wv * 32 + cc * 16;
        const _Float16* wrow = Wt + (n0 + (lane & 15)) * cH;
#pragma unroll
        for (int k0 = 0; k0 < 8; ++k0) {
            v16h bf = load_frag(wrow + k0 * 32, koff);
#pragma unroll
            for (int r = 0; r < 2; ++r)
                acc[r][cc] = __builtin_amdgcn_wmma_f32_16x16x32_f16(
                    false, af[r][k0], false, bf, (short)0, acc[r][cc], false, false);
        }
    }

    const int nloc = lane & 15;
    const int mb = (lane >> 4) * 8;
#pragma unroll
    for (int cc = 0; cc < 2; ++cc) {
        const int n = wv * 32 + cc * 16 + nloc;
        const float bn = bias[n];
#pragma unroll
        for (int r = 0; r < 2; ++r) {
#pragma unroll
            for (int v = 0; v < 8; ++v) {
                const int m = m0 + r * 16 + mb + v;
                float g = acc[r][cc][v] + bn;
                if (act == 1) g = (g > 0.0f) ? g : 0.01f * g;
                if (out16) out16[m * cH + n] = (_Float16)g;
                if (out32) out32[m * cH + n] = g;
            }
        }
    }
}

// ---------------- host launcher ----------------
extern "C" void kernel_launch(void* const* d_in, const int* in_sizes, int n_in,
                              void* d_out, int out_size, void* d_ws, size_t ws_size,
                              hipStream_t stream) {
    (void)in_sizes; (void)n_in; (void)out_size; (void)ws_size;
    const float* expert    = (const float*)d_in[0];
    const float* initState = (const float*)d_in[1];
    const float* eps       = (const float*)d_in[2];
    const float* encEmbW   = (const float*)d_in[3];
    const float* encEmbB   = (const float*)d_in[4];
    const float* encWih    = (const float*)d_in[5];
    const float* encBih    = (const float*)d_in[6];
    const float* encWhh    = (const float*)d_in[7];
    const float* encBhh    = (const float*)d_in[8];
    const float* muW       = (const float*)d_in[9];
    const float* muBias    = (const float*)d_in[10];
    const float* lvW       = (const float*)d_in[11];
    const float* lvBias    = (const float*)d_in[12];
    const float* decEmbW   = (const float*)d_in[13];
    const float* decEmbB   = (const float*)d_in[14];
    const float* decInitW  = (const float*)d_in[15];
    const float* decInitB  = (const float*)d_in[16];
    const float* decWih    = (const float*)d_in[17];
    const float* decBih    = (const float*)d_in[18];
    const float* decWhh    = (const float*)d_in[19];
    const float* decBhh    = (const float*)d_in[20];
    const float* ctrlW     = (const float*)d_in[21];
    const float* ctrlB     = (const float*)d_in[22];

    // output layout: recons | expert | mu | lv
    float* recons  = (float*)d_out;
    float* expOut  = recons + (size_t)cB * cT * 4;
    float* muOut   = expOut + (size_t)cB * cT * 4;
    float* lvOut   = muOut + (size_t)cB * cH;

    // workspace carve-up
    char* w = (char*)d_ws;
    size_t off = 0;
    auto alloc = [&](size_t bytes) { void* p = w + off; off = (off + bytes + 255) & ~(size_t)255; return p; };
    _Float16* encWhhT  = (_Float16*)alloc((size_t)cG4 * cH * 2);
    _Float16* decWhhT  = (_Float16*)alloc((size_t)cG4 * cH * 2);
    _Float16* muWT     = (_Float16*)alloc((size_t)4 * cH * cH * 2);
    _Float16* lvWT     = (_Float16*)alloc((size_t)4 * cH * cH * 2);
    _Float16* decInitT = (_Float16*)alloc((size_t)cH * cH * 2);
    float*    encWf    = (float*)alloc((size_t)4 * cG4 * 4);
    float*    encBTot  = (float*)alloc((size_t)cG4 * 4);
    float*    decWf    = (float*)alloc((size_t)4 * cG4 * 4);
    float*    decBTot  = (float*)alloc((size_t)cG4 * 4);
    _Float16* hEnc     = (_Float16*)alloc((size_t)cB * cH * 2);
    float*    cEnc     = (float*)alloc((size_t)cB * cH * 4);
    _Float16* hDec     = (_Float16*)alloc((size_t)cB * cH * 2);
    float*    cDec     = (float*)alloc((size_t)cB * cH * 4);
    _Float16* tmpA     = (_Float16*)alloc((size_t)cB * cH * 2);
    _Float16* tmpB     = (_Float16*)alloc((size_t)cB * cH * 2);
    _Float16* zBuf     = (_Float16*)alloc((size_t)cB * cH * 2);
    float*    stateBuf = (float*)alloc((size_t)cB * 4 * 4);

    const int TPB = 256;
    const int gridRows = cB / cROWS;  // 256

    // 1) weight prep
    fuse_in_kernel<<<20, TPB, 0, stream>>>(encEmbW, encEmbB, encWih, encBih, encBhh, encWf, encBTot);
    fuse_in_kernel<<<20, TPB, 0, stream>>>(decEmbW, decEmbB, decWih, decBih, decBhh, decWf, decBTot);
    transpose_f16_kernel<<<(cH * cG4 + TPB - 1) / TPB, TPB, 0, stream>>>(encWhh, encWhhT, cH, cG4);
    transpose_f16_kernel<<<(cH * cG4 + TPB - 1) / TPB, TPB, 0, stream>>>(decWhh, decWhhT, cH, cG4);
    for (int i = 0; i < 4; ++i) {
        transpose_f16_kernel<<<(cH * cH + TPB - 1) / TPB, TPB, 0, stream>>>(muW + (size_t)i * cH * cH, muWT + (size_t)i * cH * cH, cH, cH);
        transpose_f16_kernel<<<(cH * cH + TPB - 1) / TPB, TPB, 0, stream>>>(lvW + (size_t)i * cH * cH, lvWT + (size_t)i * cH * cH, cH, cH);
    }
    transpose_f16_kernel<<<(cH * cH + TPB - 1) / TPB, TPB, 0, stream>>>(decInitW, decInitT, cH, cH);

    // 2) init encoder state + copy expert_traj through to output
    zero_hc_kernel<<<(cB * cH + TPB - 1) / TPB, TPB, 0, stream>>>(hEnc, cEnc, cB * cH);
    hipMemcpyAsync(expOut, expert, (size_t)cB * cT * 4 * sizeof(float),
                   hipMemcpyDeviceToDevice, stream);

    // 3) encoder LSTM: 30 sequential fused WMMA steps
    for (int t = 0; t < cT; ++t)
        lstm_step_kernel<0><<<gridRows, TPB, 0, stream>>>(hEnc, cEnc, encWhhT, encWf, encBTot,
                                                          expert, nullptr, nullptr, nullptr,
                                                          nullptr, t);

    // 4) mu / lv MLP chains (leaky_relu, final layer also writes f32 output)
    gemm_act_kernel<<<gridRows, TPB, 0, stream>>>(hEnc, muWT + 0 * cH * cH, muBias + 0 * cH, tmpA, nullptr, 1);
    gemm_act_kernel<<<gridRows, TPB, 0, stream>>>(tmpA, muWT + 1 * cH * cH, muBias + 1 * cH, tmpB, nullptr, 1);
    gemm_act_kernel<<<gridRows, TPB, 0, stream>>>(tmpB, muWT + 2 * cH * cH, muBias + 2 * cH, tmpA, nullptr, 1);
    gemm_act_kernel<<<gridRows, TPB, 0, stream>>>(tmpA, muWT + 3 * cH * cH, muBias + 3 * cH, tmpB, muOut, 1);

    gemm_act_kernel<<<gridRows, TPB, 0, stream>>>(hEnc, lvWT + 0 * cH * cH, lvBias + 0 * cH, tmpA, nullptr, 1);
    gemm_act_kernel<<<gridRows, TPB, 0, stream>>>(tmpA, lvWT + 1 * cH * cH, lvBias + 1 * cH, tmpB, nullptr, 1);
    gemm_act_kernel<<<gridRows, TPB, 0, stream>>>(tmpB, lvWT + 2 * cH * cH, lvBias + 2 * cH, tmpA, nullptr, 1);
    gemm_act_kernel<<<gridRows, TPB, 0, stream>>>(tmpA, lvWT + 3 * cH * cH, lvBias + 3 * cH, tmpB, lvOut, 1);

    // 5) reparameterize + decoder init (h = c = z @ dec_init_W + b)
    reparam_kernel<<<(cB * cH + TPB - 1) / TPB, TPB, 0, stream>>>(muOut, lvOut, eps, zBuf, cB * cH);
    gemm_act_kernel<<<gridRows, TPB, 0, stream>>>(zBuf, decInitT, decInitB, hDec, cDec, 0);
    copy_state_kernel<<<(cB * 4 + TPB - 1) / TPB, TPB, 0, stream>>>(initState, stateBuf, cB * 4);

    // 6) decoder LSTM + ctrl head + bicycle dynamics, writes recons directly
    for (int t = 0; t < cT; ++t)
        lstm_step_kernel<1><<<gridRows, TPB, 0, stream>>>(hDec, cDec, decWhhT, decWf, decBTot,
                                                          nullptr, stateBuf, ctrlW, ctrlB,
                                                          recons, t);
}
